// RelativePositionalEncoding_24395414241946
// MI455X (gfx1250) — matrix operand order repaired and checked by reading, not compiled, not run on hardware
//
#include <hip/hip_runtime.h>
#include <math.h>

typedef __attribute__((ext_vector_type(16))) _Float16 v16h;
typedef __attribute__((ext_vector_type(8)))  float    v8f;
typedef __attribute__((ext_vector_type(4)))  unsigned int uint4v;
typedef __attribute__((ext_vector_type(8)))  unsigned int uint8v;

#define WMMA_F16(A, B, C) \
  __builtin_amdgcn_wmma_f32_16x16x32_f16(false, (A), false, (B), (short)0, (C), false, false)

// Problem constants (match reference: B=2,H=8,S=2048,D=64, max_rel=128)
constexpr int B_ = 2, H_ = 8, S_ = 2048, D_ = 64;
constexpr int MAXREL = 128;
constexpr int NREL   = 2 * MAXREL + 1;   // 257
constexpr int RELP   = 272;              // padded to 17*16 columns
constexpr int WAVES  = 4;
constexpr int KVPITCH = 68;              // 64 dwords + 4 dword TDM pad -> bank-conflict-free
// softmax done in exp2 domain with 1/sqrt(D) folded in: log2(e)/8
constexpr float SMSCALE = 0.18033688011112042f;

// Issue a TDM 2D tile load: 32 rows x 64 f32 (row stride 64 f32 in memory),
// padded +4 dwords per 64-dword row in LDS (pitch 68 f32).
// D# group0 (4 SGPRs) + group1 (8 SGPRs); 2D tensor -> VADDR2/3 omitted (NULL).
__device__ __forceinline__ void tdm_load_tile(const float* gsrc, unsigned int lds_byte_off) {
  unsigned long long ga = (unsigned long long)(uintptr_t)gsrc;
  uint4v g0;
  g0.x = 1u;                                   // count=1, is_restore=0, gather off
  g0.y = lds_byte_off;                         // lds_addr
  g0.z = (unsigned int)ga;                     // global_addr[31:0]
  g0.w = (unsigned int)((ga >> 32) & 0x01FFFFFFull) | (2u << 30);  // addr[56:32] | type=2
  uint8v g1;
  g1[0] = 0x00020000u      // data_size = 4B (code 2)
        | (1u << 20)       // pad_enable
        | (5u << 22)       // pad_interval: 64 dwords
        | (3u << 25);      // pad_amount: 4 dwords
  g1[1] = (64u << 16);     // tensor_dim0 = 64  (bits 79:48)
  g1[2] = (32u << 16);     // tensor_dim1 = 32  (bits 111:80)
  g1[3] = (64u << 16);     // tile_dim0 = 64    (bits 127:112)
  g1[4] = 32u;             // tile_dim1 = 32    (bits 143:128), tile_dim2 = 0
  g1[5] = 64u;             // tensor_dim0_stride = 64 (bits 207:160)
  g1[6] = 0u;
  g1[7] = 0u;
  asm volatile("tensor_load_to_lds %0, %1" :: "s"(g0), "s"(g1) : "memory");
}

__global__ __launch_bounds__(128)
void relattn_flash_kernel(const float* __restrict__ q,
                          const float* __restrict__ k,
                          const float* __restrict__ v,
                          const float* __restrict__ rel,
                          float* __restrict__ out)
{
  // per-wave scratch: rel-bias tile (16 x 272 f16) + P staging (16 x 32 f16)
  __shared__ _Float16 relbuf[WAVES * 16 * RELP];            // 34816 B
  __shared__ _Float16 pbuf[WAVES * 16 * 32];                //  4096 B
  // per-block TDM staging of K / V tiles (32 x 64 f32, pitch 68)
  __shared__ __align__(16) float kstage[32 * KVPITCH];      //  8704 B
  __shared__ __align__(16) float vstage[32 * KVPITCH];      //  8704 B

  const int lane  = threadIdx.x & 31;
  const int wave  = threadIdx.x >> 5;
  const int nqblk = S_ / 64;
  const int bh    = blockIdx.x / nqblk;
  const int qb    = (blockIdx.x % nqblk) * 64 + wave * 16;   // this wave's query base

  const float* Q = q + (size_t)bh * S_ * D_;
  const float* K = k + (size_t)bh * S_ * D_;
  const float* V = v + (size_t)bh * S_ * D_;

  const int lo    = lane & 15;
  const int hi    = lane >> 4;   // 0 or 1 (wave half)
  const int mbase = hi * 8;      // C-layout row base for this half

  // LDS byte offsets for TDM descriptors (generic VA low 32 bits == LDS offset)
  const unsigned int k_lds = (unsigned int)(uintptr_t)(void*)kstage;
  const unsigned int v_lds = (unsigned int)(uintptr_t)(void*)vstage;

  // ---- Q A-fragments: 16x32 f16, two frags over D (0..31, 32..63) ----
  // A layout: lane row = lane%16; lanes 0-15 hold K 0..15, lanes 16-31 K 16..31
  const float* qrow = Q + (size_t)(qb + lo) * D_;
  v16h qa0, qa1;
  {
    const int c1 = hi * 16;
#pragma unroll
    for (int j = 0; j < 8; ++j) {
      qa0[j]     = (_Float16)qrow[c1 + j];
      qa0[8 + j] = (_Float16)qrow[c1 + 8 + j];
      qa1[j]     = (_Float16)qrow[32 + c1 + j];
      qa1[8 + j] = (_Float16)qrow[32 + c1 + 8 + j];
    }
  }

  _Float16* relw = relbuf + wave * 16 * RELP;
  _Float16* pw   = pbuf   + wave * 16 * 32;

  // ---- precompute rel bias rows: relw[m][j] = Q[m,:] . rel_table[j,:] via WMMA ----
#pragma unroll 1
  for (int jt = 0; jt < RELP / 16; ++jt) {
    int jrow = jt * 16 + lo;
    if (jrow > NREL - 1) jrow = NREL - 1;        // clamp pad columns (never gathered)
    const float* rrow = rel + (size_t)jrow * D_;
    const int ds = hi * 16;                      // B layout: lanes 0-15 K 0..15, 16-31 K 16..31
    v16h b0, b1;
#pragma unroll
    for (int j = 0; j < 16; ++j) {
      b0[j] = (_Float16)rrow[ds + j];
      b1[j] = (_Float16)rrow[32 + ds + j];
    }
    v8f c = {};
    c = WMMA_F16(qa0, b0, c);
    c = WMMA_F16(qa1, b1, c);
#pragma unroll
    for (int r = 0; r < 8; ++r)
      relw[(mbase + r) * RELP + jt * 16 + lo] = (_Float16)c[r];
  }
  asm volatile("s_wait_dscnt 0" ::: "memory");   // rel table visible to this wave's gathers

  // ---- flash-attention state ----
  float mrow[8], lrow[8];
  v8f vzero = {};
  v8f acc[4];
#pragma unroll
  for (int r = 0; r < 8; ++r) { mrow[r] = -3.0e38f; lrow[r] = 0.0f; }
#pragma unroll
  for (int dt = 0; dt < 4; ++dt) acc[dt] = vzero;

  // ---- main loop over 32-key tiles, staged per-block via the Tensor Data Mover ----
#pragma unroll 1
  for (int kb = 0; kb < S_; kb += 32) {
    if (wave == 0) {
      tdm_load_tile(K + (size_t)kb * D_, k_lds);
      tdm_load_tile(V + (size_t)kb * D_, v_lds);
      __builtin_amdgcn_s_wait_tensorcnt(0);
    }
    __syncthreads();   // K/V tile visible to all waves

    // K^T B-fragments from LDS: B[d][key]; lane col = key, contiguous 16 d per lane
    v8f s0 = {}, s1 = {};
    {
      const float* kr0 = kstage + (size_t)lo * KVPITCH;
      const float* kr1 = kstage + (size_t)(16 + lo) * KVPITCH;
      const int ds = hi * 16;
      v16h kb00, kb01, kb10, kb11;
#pragma unroll
      for (int j = 0; j < 16; ++j) {
        kb00[j] = (_Float16)kr0[ds + j];
        kb01[j] = (_Float16)kr0[32 + ds + j];
        kb10[j] = (_Float16)kr1[ds + j];
        kb11[j] = (_Float16)kr1[32 + ds + j];
      }
      s0 = WMMA_F16(qa0, kb00, s0);
      s0 = WMMA_F16(qa1, kb01, s0);
      s1 = WMMA_F16(qa0, kb10, s1);
      s1 = WMMA_F16(qa1, kb11, s1);
    }

    // add relative-position bias: gather relw[m][clip(k-q)+128] from LDS
#pragma unroll
    for (int r = 0; r < 8; ++r) {
      const int m  = mbase + r;
      const int dq = qb + m;
      int i0 = (kb + lo) - dq;
      int i1 = i0 + 16;
      i0 = min(max(i0, -MAXREL), MAXREL) + MAXREL;
      i1 = min(max(i1, -MAXREL), MAXREL) + MAXREL;
      s0[r] += (float)relw[m * RELP + i0];
      s1[r] += (float)relw[m * RELP + i1];
    }

    // online softmax: row max over 32 keys (two halves + 16-lane butterfly)
    float rmax[8];
#pragma unroll
    for (int r = 0; r < 8; ++r) rmax[r] = fmaxf(s0[r], s1[r]);
#pragma unroll
    for (int off = 1; off < 16; off <<= 1)
#pragma unroll
      for (int r = 0; r < 8; ++r)
        rmax[r] = fmaxf(rmax[r], __shfl_xor(rmax[r], off, 32));

    float psum[8];
#pragma unroll
    for (int r = 0; r < 8; ++r) {
      const float mn = fmaxf(mrow[r], rmax[r]);
      const float al = exp2f((mrow[r] - mn) * SMSCALE);
      mrow[r] = mn;
      lrow[r] *= al;
#pragma unroll
      for (int dt = 0; dt < 4; ++dt) acc[dt][r] *= al;
      const float p0 = exp2f((s0[r] - mn) * SMSCALE);
      const float p1 = exp2f((s1[r] - mn) * SMSCALE);
      s0[r] = p0;
      s1[r] = p1;
      psum[r] = p0 + p1;
    }
#pragma unroll
    for (int off = 1; off < 16; off <<= 1)
#pragma unroll
      for (int r = 0; r < 8; ++r)
        psum[r] += __shfl_xor(psum[r], off, 32);
#pragma unroll
    for (int r = 0; r < 8; ++r) lrow[r] += psum[r];

    // P: C-layout f32 -> LDS row-major f16 -> A-fragment (16x32 over keys)
#pragma unroll
    for (int r = 0; r < 8; ++r) {
      const int m = mbase + r;
      pw[m * 32 + lo]      = (_Float16)s0[r];
      pw[m * 32 + 16 + lo] = (_Float16)s1[r];
    }
    asm volatile("s_wait_dscnt 0" ::: "memory");
    v16h pa;
    {
      const _Float16* prow = pw + lo * 32;
      const int koff = hi * 16;
#pragma unroll
      for (int j = 0; j < 8; ++j) {
        pa[j]     = prow[koff + j];
        pa[8 + j] = prow[koff + 8 + j];
      }
    }

    // acc += P @ V : B[key][d] fragments from staged V, 4 output d-tiles of 16
#pragma unroll
    for (int dt = 0; dt < 4; ++dt) {
      const int dcol = dt * 16 + lo;
      const int koff = hi * 16;
      v16h vb;
#pragma unroll
      for (int e = 0; e < 16; ++e)
        vb[e] = (_Float16)vstage[(size_t)(koff + e) * KVPITCH + dcol];
      acc[dt] = WMMA_F16(pa, vb, acc[dt]);
    }

    __syncthreads();   // all waves done with staged tile before next TDM overwrite
  }

  // ---- epilogue: out = acc / l ----
  float* O = out + (size_t)bh * S_ * D_;
#pragma unroll
  for (int r = 0; r < 8; ++r) {
    const float inv = 1.0f / lrow[r];
    const int m = mbase + r;
#pragma unroll
    for (int dt = 0; dt < 4; ++dt)
      O[(size_t)(qb + m) * D_ + dt * 16 + lo] = acc[dt][r] * inv;
  }
}

extern "C" void kernel_launch(void* const* d_in, const int* in_sizes, int n_in,
                              void* d_out, int out_size, void* d_ws, size_t ws_size,
                              hipStream_t stream) {
  (void)in_sizes; (void)n_in; (void)out_size; (void)d_ws; (void)ws_size;
  const float* q   = (const float*)d_in[0];
  const float* k   = (const float*)d_in[1];
  const float* v   = (const float*)d_in[2];
  const float* rel = (const float*)d_in[3];
  float* out = (float*)d_out;

  const int blocks = B_ * H_ * (S_ / 64);   // 512 blocks of 128 threads (4 waves)
  relattn_flash_kernel<<<blocks, 128, 0, stream>>>(q, k, v, rel, out);
}
